// MultiHeadAttention_90761248899399
// MI455X (gfx1250) — matrix operand order repaired and checked by reading, not compiled, not run on hardware
//
#include <hip/hip_runtime.h>
#include <hip/hip_bf16.h>
#include <math.h>

typedef __attribute__((ext_vector_type(16))) _Float16 v16h;
typedef __attribute__((ext_vector_type(8)))  _Float16 v8h;
typedef __attribute__((ext_vector_type(8)))  float    v8f;
typedef __attribute__((ext_vector_type(4)))  float    v4f;

#define DIM   256
#define NSEQ  2048
#define BATCH 8
#define ROWS  (BATCH * NSEQ)   // 16384
#define E3    (3 * DIM)        // 768

__device__ __forceinline__ float sigmoidf_(float x) {
    return 1.0f / (1.0f + __expf(-x));
}

__device__ __forceinline__ v8f wmma_f16(v16h a, v16h b, v8f c) {
    return __builtin_amdgcn_wmma_f32_16x16x32_f16(
        /*neg_a=*/false, a, /*neg_b=*/false, b,
        /*c_mod=*/(short)0, c, /*reuse_a=*/false, /*reuse_b=*/false);
}

// ---------------- Kernel 1: deg[row] = sum_m rel[row, m]  (one wave per row)
__global__ __launch_bounds__(256) void k_deg(const float* __restrict__ rel,
                                             float* __restrict__ deg) {
    int wid  = threadIdx.x >> 5;
    int lane = threadIdx.x & 31;
    int row  = blockIdx.x * 8 + wid;
    const float* p = rel + (size_t)row * NSEQ + lane * 4;
    float s = 0.f;
    #pragma unroll 4
    for (int j = 0; j < NSEQ / 128; ++j) {           // 16 x float4 per lane
        v4f v = *(const v4f*)(p + j * 128);
        s += (v[0] + v[1]) + (v[2] + v[3]);
    }
    #pragma unroll
    for (int m = 16; m >= 1; m >>= 1) s += __shfl_xor(s, m, 32);
    if (lane == 0) deg[row] = s;
}

// ---------------- Kernel 2: xg = f16( x * sigmoid(deg*W_d + b_d) )
__global__ __launch_bounds__(256) void k_gate(const float* __restrict__ x,
                                              const float* __restrict__ deg,
                                              const float* __restrict__ Wd,
                                              const float* __restrict__ bd,
                                              _Float16* __restrict__ xg) {
    int row = blockIdx.x;          // 0..16383
    int d   = threadIdx.x;         // 0..255
    size_t idx = (size_t)row * DIM + d;
    float g = sigmoidf_(deg[row] * Wd[d] + bd[d]);
    xg[idx] = (_Float16)(x[idx] * g);
}

// ---------------- Kernel 2b: W_qkv fp32 -> f16
__global__ __launch_bounds__(256) void k_cvtW(const float* __restrict__ W,
                                              _Float16* __restrict__ wq) {
    size_t idx = (size_t)blockIdx.x * 256 + threadIdx.x;   // 768*256 total
    wq[idx] = (_Float16)W[idx];
}

// ---------------- Kernel 3: QKV GEMM  [16384,256] x [256,768] + bias
// One wave per 16x16 output tile; K=256 in 8 WMMA steps.
// Routes: e<256 -> sigmoid -> qk(f16, row-major); 256..511 -> vT(f16, transposed);
//         512..767 -> res(f32, row-major)
__global__ __launch_bounds__(256) void k_qkv(const _Float16* __restrict__ xg,
                                             const _Float16* __restrict__ wq,
                                             const float* __restrict__ bqkv,
                                             _Float16* __restrict__ qk,
                                             _Float16* __restrict__ vT,
                                             float* __restrict__ res) {
    int wid  = threadIdx.x >> 5;
    int lane = threadIdx.x & 31;
    int l16  = lane & 15;
    int ph   = lane >> 4;          // which half of the wave
    int gw    = blockIdx.x * 8 + wid;
    int tileM = gw / (E3 / 16);
    int tileE = gw % (E3 / 16);
    int row0 = tileM * 16, e0 = tileE * 16;

    v8f acc = {};
    const _Float16* aBase = xg + (size_t)(row0 + l16) * DIM + ph * 8;
    const _Float16* bBase = wq + (size_t)(e0  + l16) * DIM + ph * 8;
    #pragma unroll
    for (int kc = 0; kc < 8; ++kc) {
        int k0 = kc * 32;
        v8h alo = *(const v8h*)(aBase + k0);
        v8h ahi = *(const v8h*)(aBase + k0 + 16);
        v8h blo = *(const v8h*)(bBase + k0);
        v8h bhi = *(const v8h*)(bBase + k0 + 16);
        v16h a, b;
        #pragma unroll
        for (int i = 0; i < 8; ++i) {
            a[i] = alo[i]; a[8 + i] = ahi[i];
            b[i] = blo[i]; b[8 + i] = bhi[i];
        }
        acc = wmma_f16(a, b, acc);
    }

    int e = e0 + l16;
    float bias = bqkv[e];
    #pragma unroll
    for (int r = 0; r < 8; ++r) {
        int row = row0 + r + 8 * ph;
        float v = acc[r] + bias;
        if (e < DIM) {
            qk[(size_t)row * DIM + e] = (_Float16)sigmoidf_(v);
        } else if (e < 2 * DIM) {
            vT[(size_t)(e - DIM) * ROWS + row] = (_Float16)v;
        } else {
            res[(size_t)row * DIM + (e - 2 * DIM)] = v;
        }
    }
}

// ---------------- Kernel 4: fused attention (linear normalization -> flash-style)
// Wave owns 16 rows x 128 output cols (2-way column split => only 2x redundant
// score computation). Loop m in chunks of 32:
//   S(16x32) = qk_rows . qk_m^T  (16 WMMA), * scale * rel_pos, accumulate rowsum,
//   LDS round-trip to A-layout, O(16x128) += P @ V (8 WMMA).
// Finally O /= (rowsum + 1e-6), + res, relu.
__global__ __launch_bounds__(256) void k_attn(const float* __restrict__ rel,
                                              const _Float16* __restrict__ qk,
                                              const _Float16* __restrict__ vT,
                                              const float* __restrict__ res,
                                              float* __restrict__ out) {
    __shared__ _Float16 smem[8 * 512];   // 1KB per wave (16x32 f16 P-tile)
    int wid  = threadIdx.x >> 5;
    int lane = threadIdx.x & 31;
    int l16  = lane & 15;
    int ph   = lane >> 4;

    int gw = blockIdx.x * 8 + wid;       // 2048 waves: b(8) x rowTile(128) x colSlice(2)
    int b        = gw >> 8;
    int rem      = gw & 255;
    int rowTile  = rem >> 1;
    int colSlice = rem & 1;
    int row0 = rowTile * 16;
    int col0 = colSlice * 128;
    size_t gbase = (size_t)b * NSEQ;     // global row offset of this batch

    // Preload A-fragments for the 16 qk rows (16x256 f16 = 8 fragments)
    v16h Aq[8];
    {
        const _Float16* ap = qk + (gbase + row0 + l16) * DIM + ph * 8;
        #pragma unroll
        for (int kc = 0; kc < 8; ++kc) {
            v8h lo = *(const v8h*)(ap + kc * 32);
            v8h hi = *(const v8h*)(ap + kc * 32 + 16);
            #pragma unroll
            for (int i = 0; i < 8; ++i) { Aq[kc][i] = lo[i]; Aq[kc][8 + i] = hi[i]; }
        }
    }

    v8f accO[8] = {};
    float psum[8] = {0.f, 0.f, 0.f, 0.f, 0.f, 0.f, 0.f, 0.f};
    const float kScale = 0.17677669529663687f;   // 1/sqrt(32)
    _Float16* ws = smem + wid * 512;

    for (int m0 = 0; m0 < NSEQ; m0 += 32) {
        // ---- scores: two 16x16 tiles over K=256
        v8f s0 = {}, s1 = {};
        const _Float16* bp0 = qk + (gbase + m0 + l16) * DIM + ph * 8;
        const _Float16* bp1 = bp0 + (size_t)16 * DIM;
        if (m0 + 32 < NSEQ) {   // prefetch next chunk's qk tiles (gfx1250 global_prefetch_b8)
            __builtin_prefetch(bp0 + (size_t)32 * DIM, 0, 3);
            __builtin_prefetch(bp1 + (size_t)32 * DIM, 0, 3);
        }
        #pragma unroll
        for (int kc = 0; kc < 8; ++kc) {
            v8h b0lo = *(const v8h*)(bp0 + kc * 32);
            v8h b0hi = *(const v8h*)(bp0 + kc * 32 + 16);
            v8h b1lo = *(const v8h*)(bp1 + kc * 32);
            v8h b1hi = *(const v8h*)(bp1 + kc * 32 + 16);
            v16h b0, b1;
            #pragma unroll
            for (int i = 0; i < 8; ++i) {
                b0[i] = b0lo[i]; b0[8 + i] = b0hi[i];
                b1[i] = b1lo[i]; b1[8 + i] = b1hi[i];
            }
            s0 = wmma_f16(Aq[kc], b0, s0);
            s1 = wmma_f16(Aq[kc], b1, s1);
        }

        // ---- p = s * scale * rel_pos; rowsum; stage P-tile to LDS (C->A transpose)
        #pragma unroll
        for (int r = 0; r < 8; ++r) {
            int row = row0 + r + 8 * ph;
            const float* rp = rel + (gbase + row) * (size_t)NSEQ + m0 + l16;
            __builtin_prefetch(rp + 32, 0, 3);   // next chunk's rel_pos row segment
            float p0 = s0[r] * kScale * rp[0];
            float p1 = s1[r] * kScale * rp[16];
            psum[r] += p0 + p1;
            ws[(r + 8 * ph) * 32 + l16]      = (_Float16)p0;
            ws[(r + 8 * ph) * 32 + 16 + l16] = (_Float16)p1;
        }
        __asm__ volatile("s_wait_dscnt 0" ::: "memory");  // wave's LDS stores visible

        // ---- read P back in A-fragment layout (16x32)
        v16h Pa;
        {
            const _Float16* pp = ws + l16 * 32 + ph * 8;
            v8h lo = *(const v8h*)(pp);
            v8h hi = *(const v8h*)(pp + 16);
            #pragma unroll
            for (int i = 0; i < 8; ++i) { Pa[i] = lo[i]; Pa[8 + i] = hi[i]; }
        }
        __asm__ volatile("" ::: "memory");

        // ---- O += P @ V  (V columns contiguous thanks to vT layout)
        #pragma unroll
        for (int t = 0; t < 8; ++t) {
            const _Float16* vp = vT + (size_t)(col0 + t * 16 + l16) * ROWS
                                    + gbase + m0 + ph * 8;
            v8h lo = *(const v8h*)(vp);
            v8h hi = *(const v8h*)(vp + 16);
            v16h Vb;
            #pragma unroll
            for (int i = 0; i < 8; ++i) { Vb[i] = lo[i]; Vb[8 + i] = hi[i]; }
            accO[t] = wmma_f16(Pa, Vb, accO[t]);
        }
    }

    // ---- reduce rowsums across the 16 lanes of each half (masks<16 stay in half)
    #pragma unroll
    for (int r = 0; r < 8; ++r) {
        float s = psum[r];
        #pragma unroll
        for (int m = 8; m >= 1; m >>= 1) s += __shfl_xor(s, m, 32);
        psum[r] = 1.0f / (s + 1e-6f);
    }

    // ---- normalize, add residual, relu, store
    #pragma unroll
    for (int t = 0; t < 8; ++t) {
        int col = col0 + t * 16 + l16;
        #pragma unroll
        for (int r = 0; r < 8; ++r) {
            size_t row = gbase + row0 + r + 8 * ph;
            float v = accO[t][r] * psum[r] + res[row * DIM + col];
            out[row * DIM + col] = fmaxf(v, 0.0f);
        }
    }
}

extern "C" void kernel_launch(void* const* d_in, const int* in_sizes, int n_in,
                              void* d_out, int out_size, void* d_ws, size_t ws_size,
                              hipStream_t stream) {
    (void)in_sizes; (void)n_in; (void)out_size; (void)ws_size;
    const float* x    = (const float*)d_in[0];
    const float* rel  = (const float*)d_in[1];
    const float* Wqkv = (const float*)d_in[2];
    const float* bqkv = (const float*)d_in[3];
    const float* Wd   = (const float*)d_in[4];
    const float* bd   = (const float*)d_in[5];
    float* out = (float*)d_out;

    // workspace layout (~40.4 MB total)
    char* ws = (char*)d_ws;
    float*    deg = (float*)   (ws + 0);         //   64 KB
    _Float16* xg  = (_Float16*)(ws + 65536);     //    8 MB
    _Float16* wq  = (_Float16*)(ws + 8454144);   //  384 KB
    _Float16* qk  = (_Float16*)(ws + 8847360);   //    8 MB (sigmoid applied)
    _Float16* vT  = (_Float16*)(ws + 17235968);  //    8 MB (transposed)
    float*    res = (float*)   (ws + 25624576);  //   16 MB

    k_deg <<<ROWS / 8, 256, 0, stream>>>(rel, deg);
    k_gate<<<ROWS,     256, 0, stream>>>(x, deg, Wd, bd, xg);
    k_cvtW<<<E3,       256, 0, stream>>>(Wqkv, wq);
    k_qkv <<<(ROWS / 16) * (E3 / 16) / 8, 256, 0, stream>>>(xg, wq, bqkv, qk, vT, res);
    k_attn<<<BATCH * (NSEQ / 16) * 2 / 8,  256, 0, stream>>>(rel, qk, vT, res, out);
}
